// FitLoss_36721970380859
// MI455X (gfx1250) — matrix operand order repaired and checked by reading, not compiled, not run on hardware
//
#include <hip/hip_runtime.h>
#include <hip/hip_bf16.h>
#include <math.h>
#include <stdint.h>

typedef __attribute__((ext_vector_type(16))) _Float16 v16h;
typedef __attribute__((ext_vector_type(8)))  _Float16 v8h;
typedef __attribute__((ext_vector_type(8)))  float    v8f;

#define DIM        128   // feature dimension (fixed by problem)
#define MCHUNK     128   // model rows staged per block iteration
#define NT         64    // observation columns per block (4 waves x 16)
#define THREADS    128   // 4 wave32 waves
#define LDS_STRIDE 136   // 128 + 8 f16 pad -> 4-bank rotation per row

// ---- CDNA5 async global->LDS copy (16B per lane), tracked by ASYNCcnt ----
__device__ __forceinline__ void async_b128(uint32_t lds_off, const void* g) {
  asm volatile("global_load_async_to_lds_b128 %0, %1, off"
               :: "v"(lds_off), "v"((uint64_t)(uintptr_t)g)
               : "memory");
}
__device__ __forceinline__ void wait_async0() {
#if __has_builtin(__builtin_amdgcn_s_wait_asynccnt)
  __builtin_amdgcn_s_wait_asynccnt(0);
#else
  asm volatile("s_wait_asynccnt 0x0" ::: "memory");
#endif
}
// flat LDS pointer -> wave-relative LDS byte address (aperture low 32 bits)
__device__ __forceinline__ uint32_t lds_addr(const void* p) {
  return (uint32_t)(uintptr_t)p;
}

// ============ prep: split x into f16 hi/lo, compute row terms (once) ========
__global__ void fitloss_prep(const float* __restrict__ x,
                             const float* __restrict__ xw,
                             _Float16* __restrict__ xhi,
                             _Float16* __restrict__ xlo,
                             float2* __restrict__ rowT,
                             int M)
{
  int m = blockIdx.x * blockDim.x + threadIdx.x;
  if (m >= M) return;
  const float* xr = x + (size_t)m * DIM;
  _Float16* dh = xhi + (size_t)m * DIM;
  _Float16* dl = xlo + (size_t)m * DIM;
  float sq = 0.f;
#pragma unroll
  for (int k = 0; k < DIM; k += 8) {
    float4 a = *(const float4*)(xr + k);
    float4 b = *(const float4*)(xr + k + 4);
    float f[8] = {a.x, a.y, a.z, a.w, b.x, b.y, b.z, b.w};
    v8h hh, ll;
#pragma unroll
    for (int j = 0; j < 8; ++j) {
      sq += f[j] * f[j];
      _Float16 h = (_Float16)f[j];
      hh[j] = h;
      ll[j] = (_Float16)(f[j] - (float)h);
    }
    *(v8h*)(dh + k) = hh;
    *(v8h*)(dl + k) = ll;
  }
  rowT[m] = make_float2(-0.5f * sq, __logf(xw[m]));
}

// ============ main: fused GEMM (split-f16 WMMA) + streaming logsumexp =======
__launch_bounds__(THREADS)
__global__ void fitloss_main(const _Float16* __restrict__ xhi,
                             const _Float16* __restrict__ xlo,
                             const float2* __restrict__ rowTg,
                             const float* __restrict__ y,
                             float* __restrict__ partials,
                             int M)
{
  __shared__ _Float16 ldsHi[2][MCHUNK * LDS_STRIDE];
  __shared__ _Float16 ldsLo[2][MCHUNK * LDS_STRIDE];
  __shared__ float2   rowT[2][MCHUNK];
  __shared__ float    waveSum[THREADS / 32];

  const int tid   = threadIdx.x;
  const int wid   = tid >> 5;
  const int lane  = tid & 31;
  const int col   = lane & 15;   // WMMA N index for this lane
  const int khalf = lane >> 4;   // which K-half this lane holds

  // ---- load & split this wave's 16 y columns into B fragments (f16 hi/lo) --
  const int nCol = blockIdx.x * NT + wid * 16 + col;
  const float* yrow = y + (size_t)nCol * DIM;
  float y2 = 0.f;
  v16h bhi[4], blo[4];
#pragma unroll
  for (int ks = 0; ks < 4; ++ks) {
    const int base = ks * 32 + khalf * 16;   // B layout: lane<16 holds K 0..15
#pragma unroll
    for (int i = 0; i < 16; i += 4) {
      float4 v = *(const float4*)(yrow + base + i);
      float vv[4] = {v.x, v.y, v.z, v.w};
#pragma unroll
      for (int j = 0; j < 4; ++j) {
        float f = vv[j];
        y2 += f * f;
        _Float16 h = (_Float16)f;
        bhi[ks][i + j] = h;
        blo[ks][i + j] = (_Float16)(f - (float)h);
      }
    }
  }
  y2 += __shfl_xor(y2, 16, 32);        // lanes L and L^16 cover complementary K
  const float colterm = -0.5f * y2;

  // ---- async-fill helper: copy one 128-row chunk into LDS buffer b ----
  // thread t copies model row (mc + t): 16x b128 (hi) + 16x b128 (lo);
  // threads 0..63 also copy the 1KB of row terms.
  const uint32_t hiBase  = lds_addr(&ldsHi[0][0]);
  const uint32_t loBase  = lds_addr(&ldsLo[0][0]);
  const uint32_t rtBase  = lds_addr(&rowT[0][0]);
  const uint32_t bufStep = MCHUNK * LDS_STRIDE * sizeof(_Float16); // 34816 B

#define ISSUE_FILL(mc_, b_)                                                    \
  do {                                                                         \
    const char* gh = (const char*)(xhi + (size_t)((mc_) + tid) * DIM);         \
    const char* gl = (const char*)(xlo + (size_t)((mc_) + tid) * DIM);         \
    uint32_t dh = hiBase + (b_) * bufStep + tid * (LDS_STRIDE * 2);            \
    uint32_t dl = loBase + (b_) * bufStep + tid * (LDS_STRIDE * 2);            \
    _Pragma("unroll")                                                          \
    for (int s = 0; s < 16; ++s) {                                             \
      async_b128(dh + s * 16, gh + s * 16);                                    \
      async_b128(dl + s * 16, gl + s * 16);                                    \
    }                                                                          \
    if (tid < 64)                                                              \
      async_b128(rtBase + (b_) * (MCHUNK * 8) + tid * 16,                      \
                 (const char*)(rowTg + (mc_)) + tid * 16);                     \
  } while (0)

  ISSUE_FILL(0, 0);                    // prologue: chunk 0 -> buffer 0

  float run_m = -INFINITY;             // streaming logsumexp state (per column,
  float run_s = 0.f;                   // rows split between lane and lane^16)
  int buf = 0;

  for (int mc = 0; mc < M; mc += MCHUNK) {
    wait_async0();                     // this wave's fills of `buf` complete
    __syncthreads();                   // all waves' fills visible; prev reads done
    if (mc + MCHUNK < M)               // prefetch next chunk into other buffer
      ISSUE_FILL(mc + MCHUNK, buf ^ 1);

    // ---- 8 row-tiles of 16 model rows against this wave's 16 columns ----
#pragma unroll 1
    for (int rt = 0; rt < 8; ++rt) {
      const int m0 = rt * 16;
      const _Float16* ah = &ldsHi[buf][(m0 + col) * LDS_STRIDE];
      const _Float16* al = &ldsLo[buf][(m0 + col) * LDS_STRIDE];
      v8f acc = {};
#pragma unroll
      for (int ks = 0; ks < 4; ++ks) {
        const int ko = ks * 32 + khalf * 16; // A layout: lane<16 holds K 0..15
        v8h a0 = *(const v8h*)(ah + ko);
        v8h a1 = *(const v8h*)(ah + ko + 8);
        v16h Ahi = __builtin_shufflevector(a0, a1,
                     0,1,2,3,4,5,6,7,8,9,10,11,12,13,14,15);
        v8h c0 = *(const v8h*)(al + ko);
        v8h c1 = *(const v8h*)(al + ko + 8);
        v16h Alo = __builtin_shufflevector(c0, c1,
                     0,1,2,3,4,5,6,7,8,9,10,11,12,13,14,15);
        // f32-accurate product via two-term f16 split: hi*hi + hi*lo + lo*hi
        acc = __builtin_amdgcn_wmma_f32_16x16x32_f16(false, Alo, false, bhi[ks],
                                                     (short)0, acc, false, false);
        acc = __builtin_amdgcn_wmma_f32_16x16x32_f16(false, Ahi, false, blo[ks],
                                                     (short)0, acc, false, false);
        acc = __builtin_amdgcn_wmma_f32_16x16x32_f16(false, Ahi, false, bhi[ks],
                                                     (short)0, acc, false, false);
      }
      // C_ij = log w_i - 0.5*max(d2,0); -0.5*d2 = dot - 0.5||x||^2 - 0.5||y||^2
      float c[8];
      float tmax = -INFINITY;
#pragma unroll
      for (int r = 0; r < 8; ++r) {
        float2 rr = rowT[buf][m0 + khalf * 8 + r];
        float t  = acc[r] + rr.x + colterm;
        float cr = fminf(t, 0.f) + rr.y;
        c[r] = cr;
        tmax = fmaxf(tmax, cr);
      }
      float newm = fmaxf(run_m, tmax);
      float s = run_s * __expf(run_m - newm);
#pragma unroll
      for (int r = 0; r < 8; ++r) s += __expf(c[r] - newm);
      run_m = newm;
      run_s = s;
    }
    buf ^= 1;
  }

  // merge the two lane halves (complementary model rows, same column)
  float om = __shfl_xor(run_m, 16, 32);
  float os = __shfl_xor(run_s, 16, 32);
  float fm = fmaxf(run_m, om);
  float fs = run_s * __expf(run_m - fm) + os * __expf(om - fm);
  float ll = __logf(fs) + fm;          // duplicated in lane and lane^16

  // wave sum over its 16 columns (each column counted twice -> halve)
#pragma unroll
  for (int off = 16; off > 0; off >>= 1) ll += __shfl_xor(ll, off, 32);
  ll *= 0.5f;

  if (lane == 0) waveSum[wid] = ll;
  __syncthreads();
  if (tid == 0) {
    float t = 0.f;
#pragma unroll
    for (int w = 0; w < THREADS / 32; ++w) t += waveSum[w];
    partials[blockIdx.x] = t;
  }
#undef ISSUE_FILL
}

__global__ void fitloss_reduce(const float* __restrict__ partials,
                               float* __restrict__ out, int nPart, float invN)
{
  __shared__ float sh[256];
  int t = threadIdx.x;
  float v = 0.f;
  for (int i = t; i < nPart; i += 256) v += partials[i];
  sh[t] = v;
  __syncthreads();
  for (int s = 128; s > 0; s >>= 1) {
    if (t < s) sh[t] += sh[t + s];
    __syncthreads();
  }
  if (t == 0) out[0] = -sh[0] * invN;
}

extern "C" void kernel_launch(void* const* d_in, const int* in_sizes, int n_in,
                              void* d_out, int out_size, void* d_ws, size_t ws_size,
                              hipStream_t stream)
{
  const float* x  = (const float*)d_in[0];   // (M, 128)
  const float* xw = (const float*)d_in[1];   // (M,)
  const float* y  = (const float*)d_in[2];   // (N, 128)
  float* out = (float*)d_out;

  const int M = in_sizes[1];                 // 4096
  const int N = in_sizes[2] / DIM;           // 16384

  // scratch layout: xhi | xlo | rowT | partials   (~2.1 MB total)
  char* ws = (char*)d_ws;
  size_t xBytes = (size_t)M * DIM * 2;       // f16
  _Float16* xhi     = (_Float16*)(ws);
  _Float16* xlo     = (_Float16*)(ws + xBytes);
  float2*   rowT    = (float2*)(ws + 2 * xBytes);
  float*    partials= (float*)(ws + 2 * xBytes + (size_t)M * sizeof(float2));

  const int nBlocks = N / NT;                // 256

  fitloss_prep<<<(M + 255) / 256, 256, 0, stream>>>(x, xw, xhi, xlo, rowT, M);
  fitloss_main<<<nBlocks, THREADS, 0, stream>>>(xhi, xlo, rowT, y, partials, M);
  fitloss_reduce<<<1, 256, 0, stream>>>(partials, out, nBlocks, 1.0f / (float)N);
}
